// ObjectChecklistModel_69020124447190
// MI455X (gfx1250) — compile-verified, matrix operand order checked by reading, not run on hardware
//
#include <hip/hip_runtime.h>
#include <hip/hip_bf16.h>

typedef __bf16 bf16_t;
typedef __attribute__((ext_vector_type(16))) __bf16 bf16x16;
typedef __attribute__((ext_vector_type(8)))  __bf16 bf16x8;
typedef __attribute__((ext_vector_type(8)))  float  f32x8;

// model dims
#define BB   32
#define SIN  128
#define SOUT 32
#define YY   256
#define HH   512
#define EE   256
#define ATD  32
#define DIN  1569

__device__ __forceinline__ unsigned pack_bf16_2(float a, float b)
{
    union { bf16_t h[2]; unsigned u; } p;
    p.h[0] = (bf16_t)a;
    p.h[1] = (bf16_t)b;
    return p.u;
}

union FragU { bf16x16 v; bf16x8 h[2]; unsigned u[8]; };

// ---------------------------------------------------------------------------
// Generic tiled WMMA GEMM:  C[m,n] = act( sum_k A[m,k]*B(n,k) + bias[n] )
// A: row-major MxK (f32).  B: if b_kxn==0, row-major NxK; else row-major KxN.
// REQUIRES: M % 32 == 0, N % 64 == 0, K % 32 == 0 (all call sites satisfy).
// Block = 256 threads = 8 waves (2x4 wave grid), block tile 32(M) x 64(N),
// K staged 32 at a time through LDS with f32->bf16 conversion (branchless).
// ---------------------------------------------------------------------------
__global__ __launch_bounds__(256)
void wmma_gemm_kernel(const float* __restrict__ A, int lda, long long sA,
                      const float* __restrict__ Bm, int ldb, long long sB, int b_kxn,
                      float* __restrict__ C, int ldc, long long sC,
                      const float* __restrict__ bias, int act,
                      int M, int N, int K)
{
    A  += (long long)blockIdx.z * sA;
    Bm += (long long)blockIdx.z * sB;
    C  += (long long)blockIdx.z * sC;
    const int m0 = blockIdx.y * 32;
    const int n0 = blockIdx.x * 64;
    __shared__ bf16_t lA[32][40];   // row stride 80B (16B-aligned, bank-rotating)
    __shared__ bf16_t lB[64][40];
    const int tid  = threadIdx.x;
    const int lane = tid & 31;
    const int wave = tid >> 5;
    const int wm = wave >> 2;  // 0..1
    const int wn = wave & 3;   // 0..3
    const int lm = lane & 15;
    const int hi = lane >> 4;

    const int ar = tid >> 3, ac = (tid & 7) << 2;     // A tile: float4 per thread
    const int br = tid >> 2, bc = (tid & 3) << 3;     // B tile: 8 floats per thread
    const int mrow = wm * 16 + lm;
    const int nrow = wn * 16 + lm;

    f32x8 acc = {};
    for (int k0 = 0; k0 < K; k0 += 32) {
        {   // stage A tile (32x32), vectorized + branchless
            const float4 v = *(const float4*)(A + (long long)(m0 + ar) * lda + (k0 + ac));
            unsigned* dst = (unsigned*)&lA[ar][ac];
            dst[0] = pack_bf16_2(v.x, v.y);
            dst[1] = pack_bf16_2(v.z, v.w);
        }
        {   // stage B tile (64x32), branchless
            float vb[8];
            if (b_kxn) {
                // one vector multiply for the base; uniform j*ldb steps fold to SALU
                const float* pb = Bm + (long long)(k0 + bc) * ldb + (n0 + br);
                #pragma unroll
                for (int j = 0; j < 8; ++j)
                    vb[j] = pb[(long long)j * ldb];
            } else {
                const float* pb = Bm + (long long)(n0 + br) * ldb + (k0 + bc);
                #pragma unroll
                for (int j = 0; j < 8; ++j) vb[j] = pb[j];
                if (k0 + 32 < K)
                    __builtin_prefetch(pb + 32, 0, 0);   // global_prefetch_b8 next K tile
            }
            unsigned* dst = (unsigned*)&lB[br][bc];
            #pragma unroll
            for (int j = 0; j < 4; ++j) dst[j] = pack_bf16_2(vb[2*j], vb[2*j+1]);
        }
        __syncthreads();
        // fragments: two contiguous 8-half runs per lane -> 2x ds_load_b128 each
        FragU ua, ub;
        ua.h[0] = *(const bf16x8*)&lA[mrow][8 * hi];
        ua.h[1] = *(const bf16x8*)&lA[mrow][16 + 8 * hi];
        ub.h[0] = *(const bf16x8*)&lB[nrow][8 * hi];
        ub.h[1] = *(const bf16x8*)&lB[nrow][16 + 8 * hi];
        acc = __builtin_amdgcn_wmma_f32_16x16x32_bf16(false, ua.v, false, ub.v,
                                                      (short)0, acc, false, false);
        __syncthreads();
    }
    const int nn = n0 + wn * 16 + lm;
    const float bv = bias ? bias[nn] : 0.0f;
    #pragma unroll
    for (int q = 0; q < 8; ++q) {
        const int mm = m0 + wm * 16 + q + 8 * hi;
        float r = acc[q] + bv;
        if (act == 1) r = tanhf(r);
        C[(long long)mm * ldc + nn] = r;
    }
}

// ---------------------------------------------------------------------------
// Persistent bidirectional GRU encoder scan. One workgroup (1024 thr = 32
// waves) per direction (blockIdx.z). h kept in LDS (bf16), gh exchanged via
// LDS. Per step: gh = h @ Whh^T via WMMA (2 M-tiles x 96 N-tiles, 6 tiles
// per wave), then elementwise GRU gate. gi (x@Wih^T + bih) precomputed.
// ---------------------------------------------------------------------------
__global__ __launch_bounds__(1024, 1)
void encoder_scan_kernel(const float* __restrict__ GI_f, const float* __restrict__ GI_b,
                         const float* __restrict__ Whh_f, const float* __restrict__ Whh_b,
                         const float* __restrict__ bhh_f, const float* __restrict__ bhh_b,
                         float* __restrict__ hs_f, float* __restrict__ hs_b)
{
    const bool back = (blockIdx.z == 1);
    const float* GI  = back ? GI_b  : GI_f;
    const float* Whh = back ? Whh_b : Whh_f;
    const float* bhh = back ? bhh_b : bhh_f;
    float*       hs  = back ? hs_b  : hs_f;

    __shared__ bf16_t hA[32][536];     // 32 x 512 (+24 pad; 1072B stride, 16B-aligned)
    __shared__ bf16_t ghL[32][1544];   // 32 x 1536 (+8 pad)

    const int tid  = threadIdx.x;
    const int lane = tid & 31;
    const int wave = tid >> 5;
    const int lm = lane & 15;
    const int hi = lane >> 4;

    for (int i = tid; i < 32 * 536; i += 1024) ((bf16_t*)hA)[i] = (bf16_t)0.0f;
    __syncthreads();

    for (int step = 0; step < SIN; ++step) {
        const int t = back ? (SIN - 1 - step) : step;
        f32x8 acc[2][3] = {};
        for (int k0 = 0; k0 < HH; k0 += 32) {
            FragU fa[2], fb[3];
            #pragma unroll
            for (int mt = 0; mt < 2; ++mt) {
                const int m = mt * 16 + lm;
                fa[mt].h[0] = *(const bf16x8*)&hA[m][k0 + 8 * hi];
                fa[mt].h[1] = *(const bf16x8*)&hA[m][k0 + 16 + 8 * hi];
            }
            #pragma unroll
            for (int nt = 0; nt < 3; ++nt) {
                const int n = (wave * 3 + nt) * 16 + lm;
                const float* br = Whh + (long long)n * HH + k0;
                float4 v0 = *(const float4*)(br + 8 * hi);
                float4 v1 = *(const float4*)(br + 8 * hi + 4);
                float4 v2 = *(const float4*)(br + 16 + 8 * hi);
                float4 v3 = *(const float4*)(br + 20 + 8 * hi);
                fb[nt].u[0] = pack_bf16_2(v0.x, v0.y);
                fb[nt].u[1] = pack_bf16_2(v0.z, v0.w);
                fb[nt].u[2] = pack_bf16_2(v1.x, v1.y);
                fb[nt].u[3] = pack_bf16_2(v1.z, v1.w);
                fb[nt].u[4] = pack_bf16_2(v2.x, v2.y);
                fb[nt].u[5] = pack_bf16_2(v2.z, v2.w);
                fb[nt].u[6] = pack_bf16_2(v3.x, v3.y);
                fb[nt].u[7] = pack_bf16_2(v3.z, v3.w);
            }
            #pragma unroll
            for (int mt = 0; mt < 2; ++mt)
                #pragma unroll
                for (int nt = 0; nt < 3; ++nt)
                    acc[mt][nt] = __builtin_amdgcn_wmma_f32_16x16x32_bf16(
                        false, fa[mt].v, false, fb[nt].v, (short)0, acc[mt][nt], false, false);
        }
        // scatter gh to LDS
        #pragma unroll
        for (int mt = 0; mt < 2; ++mt)
            #pragma unroll
            for (int nt = 0; nt < 3; ++nt) {
                const int n = (wave * 3 + nt) * 16 + lm;
                #pragma unroll
                for (int q = 0; q < 8; ++q) {
                    const int m = mt * 16 + q + 8 * hi;
                    ghL[m][n] = (bf16_t)acc[mt][nt][q];
                }
            }
        __syncthreads();
        // GRU gate: 32*512 elements, 16 per thread
        for (int i = tid; i < 32 * HH; i += 1024) {
            const int b = i >> 9, col = i & 511;
            const long long g = ((long long)b * SIN + t) * (3 * HH);
            float ir  = GI[g + col];
            float iz  = GI[g + HH + col];
            float in_ = GI[g + 2 * HH + col];
            float hr = (float)ghL[b][col]          + bhh[col];
            float hz = (float)ghL[b][col + HH]     + bhh[col + HH];
            float hn = (float)ghL[b][col + 2 * HH] + bhh[col + 2 * HH];
            float r = 1.0f / (1.0f + __expf(-(ir + hr)));
            float z = 1.0f / (1.0f + __expf(-(iz + hz)));
            float n = tanhf(in_ + r * hn);
            float hprev = (float)hA[b][col];
            float hnew = (1.0f - z) * n + z * hprev;
            hA[b][col] = (bf16_t)hnew;
            hs[(long long)t * (32 * HH) + i] = hnew;
        }
        __syncthreads();
    }
}

// ------------------------- small helper kernels -----------------------------

__global__ void gather_rows_kernel(const float* __restrict__ table, const int* __restrict__ idx,
                                   float* __restrict__ out, int D, long long n)
{
    long long i = (long long)blockIdx.x * 256 + threadIdx.x;
    if (i >= n) return;
    long long r = i / D;
    int c = (int)(i % D);
    out[i] = table[(long long)idx[r] * D + c];
}

__global__ void add_full_kernel(float* __restrict__ a, const float* __restrict__ b, long long n)
{
    long long i = (long long)blockIdx.x * 256 + threadIdx.x;
    if (i < n) a[i] += b[i];
}

// GI_fix[(b*Y+y)*1536+c] += atg[y*1536+c]
__global__ void add_bcast_kernel(float* __restrict__ a, const float* __restrict__ g, long long n)
{
    long long i = (long long)blockIdx.x * 256 + threadIdx.x;
    if (i >= n) return;
    long long row = i / 1536;
    int col = (int)(i % 1536);
    a[i] += g[(long long)(row & (YY - 1)) * 1536 + col];
}

// scores (b,s,y) -> softmax over s -> alphaT (b,y,s)
__global__ __launch_bounds__(256)
void softmax_alpha_kernel(const float* __restrict__ scores, float* __restrict__ alphaT)
{
    int idx = blockIdx.x * 256 + threadIdx.x;   // b*Y + y
    int b = idx >> 8, y = idx & 255;
    const float* sc = scores + (long long)b * SIN * YY + y;
    float mx = -1e30f;
    for (int s = 0; s < SIN; ++s) mx = fmaxf(mx, sc[(long long)s * YY]);
    float sum = 0.0f;
    for (int s = 0; s < SIN; ++s) sum += __expf(sc[(long long)s * YY] - mx);
    float inv = 1.0f / sum;
    float* out = alphaT + (long long)idx * SIN;
    for (int s = 0; s < SIN; ++s) out[s] = __expf(sc[(long long)s * YY] - mx) * inv;
}

__global__ void build_encout_kernel(const float* __restrict__ hs_f, const float* __restrict__ hs_b,
                                    float* __restrict__ enc_out)
{
    long long i = (long long)blockIdx.x * 256 + threadIdx.x;  // < 4096*1024
    int col = (int)(i & 1023);
    long long rowbs = i >> 10;
    int b = (int)(rowbs >> 7), s = (int)(rowbs & 127);
    float v = (col < HH) ? hs_f[((long long)s * BB + b) * HH + col]
                         : hs_b[((long long)s * BB + b) * HH + (col - HH)];
    enc_out[i] = v;
}

// hidden_cat flat f: (f<16384 ? hT_f : hT_b)[f & 16383]  (matches jnp.stack+reshape)
__global__ void build_hcat_kernel(const float* __restrict__ hTf, const float* __restrict__ hTb,
                                  float* __restrict__ hcat)
{
    int f = blockIdx.x * 256 + threadIdx.x;   // < 32768
    hcat[f] = (f < 16384 ? hTf : hTb)[f & 16383];
}

__global__ void bcast_h0_kernel(const float* __restrict__ h0, float* __restrict__ hdec)
{
    long long i = (long long)blockIdx.x * 256 + threadIdx.x;  // < 8192*512
    long long row = i >> 9;
    hdec[i] = h0[(long long)(row >> 8) * HH + (i & 511)];
}

__global__ void build_yprev_kernel(const float* __restrict__ yv, float* __restrict__ yprev)
{
    long long i = (long long)blockIdx.x * 256 + threadIdx.x;  // < 1024*256
    long long bt = i >> 8;
    int t = (int)(bt & (SOUT - 1));
    yprev[i] = (t == 0) ? 0.0f : yv[i - YY];
}

// Decoder GRU gate. gi assembled on the fly from precomputed pieces.
__global__ __launch_bounds__(256)
void dec_gate_kernel(const float* __restrict__ GI_fix, const float* __restrict__ gh,
                     const float* __restrict__ Rall, const float* __restrict__ Yall,
                     const float* __restrict__ checked, const float* __restrict__ dec_Wih,
                     const float* __restrict__ bhh, const float* __restrict__ hprev,
                     float* __restrict__ hout, int t)
{
    long long i = (long long)blockIdx.x * 256 + threadIdx.x;  // < 8192*512
    int col = (int)(i & 511);
    long long row = i >> 9;                 // b*Y + y
    int b = (int)(row >> 8), y = (int)(row & 255);
    long long g  = row * 1536;
    long long rt = ((long long)b * SOUT + t) * 1536;
    float chkv = checked[((long long)b * (SOUT + 1) + t) * YY + y];
    int c0 = col, c1 = col + HH, c2 = col + 2 * HH;
    float ir  = GI_fix[g + c0] + chkv * dec_Wih[(long long)c0 * DIN + 1024] + Rall[rt + c0] + Yall[rt + c0];
    float iz  = GI_fix[g + c1] + chkv * dec_Wih[(long long)c1 * DIN + 1024] + Rall[rt + c1] + Yall[rt + c1];
    float in_ = GI_fix[g + c2] + chkv * dec_Wih[(long long)c2 * DIN + 1024] + Rall[rt + c2] + Yall[rt + c2];
    float hr = gh[g + c0] + bhh[c0];
    float hz = gh[g + c1] + bhh[c1];
    float hn = gh[g + c2] + bhh[c2];
    float r = 1.0f / (1.0f + __expf(-(ir + hr)));
    float z = 1.0f / (1.0f + __expf(-(iz + hz)));
    float n = tanhf(in_ + r * hn);
    hout[i] = (1.0f - z) * n + z * hprev[i];
}

// logits = h2 @ dec_W^T + dec_b, log_softmax over Y per b, accumulate loss
__global__ __launch_bounds__(256)
void dec_loss_kernel(const float* __restrict__ h2, const float* __restrict__ decW,
                     const float* __restrict__ decb, const float* __restrict__ yt,
                     float* __restrict__ acc, int t)
{
    __shared__ float sh[256];
    int b = blockIdx.x, yi = threadIdx.x;
    const float4* h4 = (const float4*)(h2 + ((long long)b * YY + yi) * HH);
    const float4* w4 = (const float4*)decW;
    float s = 0.0f;
    #pragma unroll 4
    for (int k = 0; k < HH / 4; ++k) {
        float4 a = h4[k], w = w4[k];
        s += a.x * w.x + a.y * w.y + a.z * w.z + a.w * w.w;
    }
    s += decb[0];
    sh[yi] = s; __syncthreads();
    for (int off = 128; off > 0; off >>= 1) { if (yi < off) sh[yi] = fmaxf(sh[yi], sh[yi + off]); __syncthreads(); }
    float mx = sh[0]; __syncthreads();
    sh[yi] = __expf(s - mx); __syncthreads();
    for (int off = 128; off > 0; off >>= 1) { if (yi < off) sh[yi] += sh[yi + off]; __syncthreads(); }
    float lse = mx + __logf(sh[0]); __syncthreads();
    float yv = yt[((long long)b * SOUT + t) * YY + yi];
    sh[yi] = -(s - lse) * yv; __syncthreads();
    for (int off = 128; off > 0; off >>= 1) { if (yi < off) sh[yi] += sh[yi + off]; __syncthreads(); }
    if (yi == 0) atomicAdd(acc, sh[0]);
}

// chk = sigmoid(tanh(fix + at + h_part) @ W2 + b2); BCE vs checked[:,t+1]; one wave per row
__global__ __launch_bounds__(256)
void chk_loss_kernel(const float* __restrict__ chk_fix, const float* __restrict__ chk_at,
                     const float* __restrict__ chk_h, const float* __restrict__ W2,
                     const float* __restrict__ b2, const float* __restrict__ checked,
                     const float* __restrict__ yt, float* __restrict__ acc, int t)
{
    int lane = threadIdx.x & 31, wave = threadIdx.x >> 5;
    long long row = (long long)blockIdx.x * 8 + wave;   // < 8192
    int b = (int)(row >> 8), y = (int)(row & 255);
    float s = 0.0f;
    #pragma unroll
    for (int j = 0; j < 16; ++j) {
        int n = lane + j * 32;
        float v = chk_fix[row * HH + n] + chk_at[(long long)y * HH + n] + chk_h[row * HH + n];
        s += W2[n] * tanhf(v);
    }
    for (int off = 16; off > 0; off >>= 1) s += __shfl_xor(s, off, 32);
    if (lane == 0) {
        float x = s + b2[0];
        float lp = fminf(x, 0.0f)  - log1pf(__expf(-fabsf(x)));   // log(sigmoid(x))
        float ln = fminf(-x, 0.0f) - log1pf(__expf(-fabsf(x)));   // log(1-sigmoid(x))
        float cn = checked[((long long)b * (SOUT + 1) + (t + 1)) * YY + y];
        float yv = yt[((long long)b * SOUT + t) * YY + y];
        atomicAdd(acc, -(cn * lp + (1.0f - cn) * ln) * yv);
    }
}

__global__ void reduce_sum_kernel(const float* __restrict__ x, long long n, float* __restrict__ acc)
{
    __shared__ float sh[256];
    long long i = (long long)blockIdx.x * 256 + threadIdx.x;
    sh[threadIdx.x] = (i < n) ? x[i] : 0.0f;
    __syncthreads();
    for (int off = 128; off > 0; off >>= 1) { if (threadIdx.x < off) sh[threadIdx.x] += sh[threadIdx.x + off]; __syncthreads(); }
    if (threadIdx.x == 0) atomicAdd(acc, sh[0]);
}

__global__ void init_acc_kernel(float* __restrict__ acc)
{
    if (threadIdx.x < 4) acc[threadIdx.x] = 0.0f;
}

__global__ void finalize_kernel(const float* __restrict__ acc, float* __restrict__ out)
{
    if (threadIdx.x == 0) {
        float ys = acc[2];
        float l = acc[0] / ys, cl = acc[1] / ys;
        out[0] = l + cl; out[1] = l; out[2] = cl;
    }
}

// ---------------------------------------------------------------------------

extern "C" void kernel_launch(void* const* d_in, const int* in_sizes, int n_in,
                              void* d_out, int out_size, void* d_ws, size_t ws_size,
                              hipStream_t stream)
{
    (void)in_sizes; (void)n_in; (void)out_size; (void)ws_size;
    const int*   x           = (const int*)d_in[0];
    const int*   action_key  = (const int*)d_in[1];
    const int*   action_type = (const int*)d_in[2];
    const int*   cur_room    = (const int*)d_in[3];
    const float* checked     = (const float*)d_in[4];
    const float* yv          = (const float*)d_in[5];
    const float* emb_table   = (const float*)d_in[6];
    const float* atype_table = (const float*)d_in[7];
    const float* enc_Wih_f   = (const float*)d_in[8];
    const float* enc_Whh_f   = (const float*)d_in[9];
    const float* enc_bih_f   = (const float*)d_in[10];
    const float* enc_bhh_f   = (const float*)d_in[11];
    const float* enc_Wih_b   = (const float*)d_in[12];
    const float* enc_Whh_b   = (const float*)d_in[13];
    const float* enc_bih_b   = (const float*)d_in[14];
    const float* enc_bhh_b   = (const float*)d_in[15];
    const float* trans_W     = (const float*)d_in[16];
    const float* trans_b     = (const float*)d_in[17];
    const float* merge_W     = (const float*)d_in[18];
    const float* merge_b     = (const float*)d_in[19];
    const float* dec_Wih     = (const float*)d_in[20];
    const float* dec_Whh     = (const float*)d_in[21];
    const float* dec_bih     = (const float*)d_in[22];
    const float* dec_bhh     = (const float*)d_in[23];
    const float* dec_W       = (const float*)d_in[24];
    const float* dec_b       = (const float*)d_in[25];
    const float* chk_W1      = (const float*)d_in[26];
    const float* chk_b1      = (const float*)d_in[27];
    const float* chk_W2      = (const float*)d_in[28];
    const float* chk_b2      = (const float*)d_in[29];

    float* w = (float*)d_ws;
    size_t off = 0;
    auto alloc = [&](size_t n) { float* p = w + off; off += n; return p; };

    float* X_emb    = alloc((size_t)4096 * 256);
    float* act_emb  = alloc((size_t)256 * 256);
    float* room_emb = alloc((size_t)1024 * 256);
    float* AT       = alloc((size_t)256 * 32);
    float* GI_f     = alloc((size_t)4096 * 1536);
    float* GI_b     = alloc((size_t)4096 * 1536);
    float* hs_f     = alloc((size_t)128 * 32 * 512);
    float* hs_b     = alloc((size_t)128 * 32 * 512);
    float* enc_out  = alloc((size_t)4096 * 1024);
    float* Tbuf     = alloc((size_t)4096 * 256);
    float* scores   = alloc((size_t)4096 * 256);
    float* alphaT   = alloc((size_t)8192 * 128);
    float* attn     = alloc((size_t)8192 * 1024);
    float* hcat     = alloc((size_t)32 * 1024);
    float* h0       = alloc((size_t)32 * 512);
    float* GI_fix   = alloc((size_t)8192 * 1536);
    float* atg      = alloc((size_t)256 * 1536);
    float* Rall     = alloc((size_t)1024 * 1536);
    float* yprev    = alloc((size_t)1024 * 256);
    float* YEmb     = alloc((size_t)1024 * 256);
    float* Yall     = alloc((size_t)1024 * 1536);
    float* chk_fix  = alloc((size_t)8192 * 512);
    float* chk_at   = alloc((size_t)256 * 512);
    float* chk_hb   = alloc((size_t)8192 * 512);
    float* gh_dec   = alloc((size_t)8192 * 1536);
    float* hdecA    = alloc((size_t)8192 * 512);
    float* hdecB    = alloc((size_t)8192 * 512);
    float* acc      = alloc(8);

    auto gemm = [&](const float* A, int lda, long long sA,
                    const float* Bp, int ldb, long long sB, int bkxn,
                    float* C, int ldc, long long sC,
                    const float* bias, int act, int M, int N, int K, int batch) {
        dim3 g((N + 63) / 64, (M + 31) / 32, batch);
        wmma_gemm_kernel<<<g, dim3(256), 0, stream>>>(A, lda, sA, Bp, ldb, sB, bkxn,
                                                      C, ldc, sC, bias, act, M, N, K);
    };
    auto blocks = [](long long n) { return (unsigned)((n + 255) / 256); };

    init_acc_kernel<<<1, 32, 0, stream>>>(acc);

    // gathers
    gather_rows_kernel<<<blocks(4096LL * 256), 256, 0, stream>>>(emb_table, x, X_emb, 256, 4096LL * 256);
    gather_rows_kernel<<<blocks(256LL * 256), 256, 0, stream>>>(emb_table, action_key, act_emb, 256, 256LL * 256);
    gather_rows_kernel<<<blocks(1024LL * 256), 256, 0, stream>>>(emb_table, cur_room, room_emb, 256, 1024LL * 256);
    gather_rows_kernel<<<blocks(256LL * 32), 256, 0, stream>>>(atype_table, action_type, AT, 32, 256LL * 32);

    // encoder input GEMMs (gi for all steps, both directions)
    gemm(X_emb, 256, 0, enc_Wih_f, 256, 0, 0, GI_f, 1536, 0, enc_bih_f, 0, 4096, 1536, 256, 1);
    gemm(X_emb, 256, 0, enc_Wih_b, 256, 0, 0, GI_b, 1536, 0, enc_bih_b, 0, 4096, 1536, 256, 1);

    // persistent bidirectional scan (one workgroup per direction)
    encoder_scan_kernel<<<dim3(1, 1, 2), 1024, 0, stream>>>(GI_f, GI_b, enc_Whh_f, enc_Whh_b,
                                                            enc_bhh_f, enc_bhh_b, hs_f, hs_b);

    build_encout_kernel<<<blocks(4096LL * 1024), 256, 0, stream>>>(hs_f, hs_b, enc_out);

    // trans_out = tanh(enc_out @ trans_W^T + b); then (+ emb) for shared score GEMM
    gemm(enc_out, 1024, 0, trans_W, 1024, 0, 0, Tbuf, 256, 0, trans_b, 1, 4096, 256, 1024, 1);
    add_full_kernel<<<blocks(4096LL * 256), 256, 0, stream>>>(Tbuf, X_emb, 4096LL * 256);

    // scores (b,s,y) = (emb + trans) @ action_emb^T
    gemm(Tbuf, 256, 0, act_emb, 256, 0, 0, scores, 256, 0, nullptr, 0, 4096, 256, 256, 1);
    softmax_alpha_kernel<<<32, 256, 0, stream>>>(scores, alphaT);

    // attention[b] = alphaT_b (Y x SIN) @ enc_out_b (SIN x 2H), batched over b
    gemm(alphaT, 128, 32768LL, enc_out, 1024, 131072LL, 1,
         attn, 1024, 262144LL, nullptr, 0, 256, 1024, 128, 32);

    // decoder initial hidden
    build_hcat_kernel<<<128, 256, 0, stream>>>(hs_f + (size_t)127 * 16384, hs_b, hcat);
    gemm(hcat, 1024, 0, merge_W, 1024, 0, 0, h0, 512, 0, merge_b, 1, 32, 512, 1024, 1);
    bcast_h0_kernel<<<blocks(8192LL * 512), 256, 0, stream>>>(h0, hdecA);

    // step-invariant pieces of dec_in @ dec_Wih^T
    gemm(attn, 1024, 0, dec_Wih, DIN, 0, 0, GI_fix, 1536, 0, dec_bih, 0, 8192, 1536, 1024, 1);
    gemm(AT, 32, 0, dec_Wih + 1281, DIN, 0, 0, atg, 1536, 0, nullptr, 0, 256, 1536, 32, 1);
    add_bcast_kernel<<<blocks(8192LL * 1536), 256, 0, stream>>>(GI_fix, atg, 8192LL * 1536);
    gemm(room_emb, 256, 0, dec_Wih + 1025, DIN, 0, 0, Rall, 1536, 0, nullptr, 0, 1024, 1536, 256, 1);
    build_yprev_kernel<<<blocks(1024LL * 256), 256, 0, stream>>>(yv, yprev);
    gemm(yprev, 256, 0, act_emb, 256, 0, 1, YEmb, 256, 0, nullptr, 0, 1024, 256, 256, 1);
    gemm(YEmb, 256, 0, dec_Wih + 1313, DIN, 0, 0, Yall, 1536, 0, nullptr, 0, 1024, 1536, 256, 1);

    // step-invariant pieces of checklist head
    gemm(attn, 1024, 0, chk_W1, 1568, 0, 0, chk_fix, 512, 0, chk_b1, 0, 8192, 512, 1024, 1);
    gemm(AT, 32, 0, chk_W1 + 1024, 1568, 0, 0, chk_at, 512, 0, nullptr, 0, 256, 512, 32, 1);

    reduce_sum_kernel<<<blocks(32LL * 32 * 256), 256, 0, stream>>>(yv, 32LL * 32 * 256, acc + 2);

    // decoder scan
    float* hc = hdecA;
    float* hn = hdecB;
    for (int t = 0; t < SOUT; ++t) {
        gemm(hc, 512, 0, dec_Whh, 512, 0, 0, gh_dec, 1536, 0, nullptr, 0, 8192, 1536, 512, 1);
        dec_gate_kernel<<<blocks(8192LL * 512), 256, 0, stream>>>(GI_fix, gh_dec, Rall, Yall,
                                                                  checked, dec_Wih, dec_bhh, hc, hn, t);
        dec_loss_kernel<<<32, 256, 0, stream>>>(hn, dec_W, dec_b, yv, acc + 0, t);
        gemm(hn, 512, 0, chk_W1 + 1056, 1568, 0, 0, chk_hb, 512, 0, nullptr, 0, 8192, 512, 512, 1);
        chk_loss_kernel<<<1024, 256, 0, stream>>>(chk_fix, chk_at, chk_hb, chk_W2, chk_b2,
                                                  checked, yv, acc + 1, t);
        float* tmp = hc; hc = hn; hn = tmp;
    }

    finalize_kernel<<<1, 32, 0, stream>>>(acc, (float*)d_out);
}
